// MultiHeadAttention_12025908429076
// MI455X (gfx1250) — matrix operand order repaired and checked by reading, not compiled, not run on hardware
//
#include <hip/hip_runtime.h>

typedef __attribute__((ext_vector_type(16))) _Float16 v16h;
typedef __attribute__((ext_vector_type(8)))  _Float16 v8h;
typedef __attribute__((ext_vector_type(8)))  float    v8f;
typedef __attribute__((ext_vector_type(4)))  int      v4i;

union V16 { v16h v; v8h h[2]; };

static __device__ __forceinline__ v8f wmma_f16(v16h a, v16h b, v8f c) {
  return __builtin_amdgcn_wmma_f32_16x16x32_f16(false, a, false, b, (short)0, c, false, false);
}

#if defined(__has_builtin)
#if __has_builtin(__builtin_amdgcn_global_load_async_to_lds_b128)
#define HAVE_ASYNC_LDS 1
#endif
#endif
#ifndef HAVE_ASYNC_LDS
#define HAVE_ASYNC_LDS 0
#endif

#if HAVE_ASYNC_LDS
typedef v4i __attribute__((address_space(1)))* v4i_gptr;
typedef v4i __attribute__((address_space(3)))* v4i_lptr;
#endif

// Copy 16 bytes global -> LDS. Async (ASYNCcnt-tracked, no VGPR bounce) when the
// gfx1250 builtin is available, otherwise load+ds_store fallback.
static __device__ __forceinline__ void gl2lds_b128(const _Float16* g, _Float16* l) {
#if HAVE_ASYNC_LDS
  __builtin_amdgcn_global_load_async_to_lds_b128((v4i_gptr)g, (v4i_lptr)l, 0, 0);
#else
  *(v8h*)l = *(const v8h*)g;
#endif
}

static __device__ __forceinline__ void wait_gl2lds() {
#if HAVE_ASYNC_LDS
#if __has_builtin(__builtin_amdgcn_s_wait_asynccnt)
  __builtin_amdgcn_s_wait_asynccnt(0);
#else
  asm volatile("s_wait_asynccnt 0x0" ::: "memory");
#endif
#endif
}

#define B_ 2
#define S_ 2048
#define E_ 1024
#define H_ 16
#define D_ 64

// ---------------- conversion / layout kernels ----------------

__global__ void cvt_f32_to_f16(const float* __restrict__ src, _Float16* __restrict__ dst, int n) {
  for (int i = blockIdx.x * blockDim.x + threadIdx.x; i < n; i += gridDim.x * blockDim.x)
    dst[i] = (_Float16)src[i];
}

// W: [G][K][N] f32  ->  Wt: [G][N][K] f16   (makes WMMA B-fragment columns contiguous)
__global__ void transpose_to_f16(const float* __restrict__ W, _Float16* __restrict__ Wt,
                                 int G, int K, int N) {
  int total = G * K * N;
  for (int i = blockIdx.x * blockDim.x + threadIdx.x; i < total; i += gridDim.x * blockDim.x) {
    int g = i / (K * N);
    int r = i - g * K * N;
    int k = r / N;
    int n = r - k * N;
    Wt[((long)g * N + n) * K + k] = (_Float16)W[i];
  }
}

// ---------------- QKV projection: q/k/v = x @ W ----------------
__global__ __launch_bounds__(128) void qkv_gemm(
    const _Float16* __restrict__ xh,   // [B*S][E]
    const _Float16* __restrict__ wqt,  // [H*D][E]
    const _Float16* __restrict__ wkt,  // [H*D][E]
    const _Float16* __restrict__ wvt,  // [H*D][E]
    _Float16* __restrict__ qo,         // [B*H][S][D]
    _Float16* __restrict__ ko,         // [B*H][S][D]
    _Float16* __restrict__ vto)        // [B*H][D][S]  (transposed for P@V later)
{
  __shared__ _Float16 As[64 * 40];     // 64x32 chunk, padded stride 40 halves
  const int tid  = threadIdx.x;
  const int wave = tid >> 5, lane = tid & 31;
  const int c = lane >> 4, mn = lane & 15;
  const int mt = blockIdx.x & 31;      // S/64 = 32 tiles
  const int bh = blockIdx.x >> 5;
  const int b  = bh >> 4, h = bh & 15;

  const _Float16* xbase = xh + ((long)b * S_ + mt * 64) * E_;

  v8f zf = {};
  v8f aq[4], ak[4], av[4];
#pragma unroll
  for (int t = 0; t < 4; ++t) { aq[t] = zf; ak[t] = zf; av[t] = zf; }

  const int arow = tid >> 1;
  const int acol = (tid & 1) * 16;

  for (int k0 = 0; k0 < E_; k0 += 32) {
    __syncthreads();
    gl2lds_b128(&xbase[(long)arow * E_ + k0 + acol],     &As[arow * 40 + acol]);
    gl2lds_b128(&xbase[(long)arow * E_ + k0 + acol + 8], &As[arow * 40 + acol + 8]);
    wait_gl2lds();
    __syncthreads();

    // A fragment: lane(c,m) e -> K=16c+e (contiguous)
    V16 a;
    a.h[0] = *(const v8h*)&As[(wave * 16 + mn) * 40 + c * 16];
    a.h[1] = *(const v8h*)&As[(wave * 16 + mn) * 40 + c * 16 + 8];

#pragma unroll
    for (int t = 0; t < 4; ++t) {
      const long col = (long)h * D_ + t * 16 + mn;
      const _Float16* wq = wqt + col * E_ + k0 + c * 16;   // B: K = e + 16c, contiguous
      const _Float16* wk = wkt + col * E_ + k0 + c * 16;
      const _Float16* wv = wvt + col * E_ + k0 + c * 16;
      V16 bq, bk, bv;
      bq.h[0] = *(const v8h*)&wq[0]; bq.h[1] = *(const v8h*)&wq[8];
      bk.h[0] = *(const v8h*)&wk[0]; bk.h[1] = *(const v8h*)&wk[8];
      bv.h[0] = *(const v8h*)&wv[0]; bv.h[1] = *(const v8h*)&wv[8];
      aq[t] = wmma_f16(a.v, bq.v, aq[t]);
      ak[t] = wmma_f16(a.v, bk.v, ak[t]);
      av[t] = wmma_f16(a.v, bv.v, av[t]);
    }
  }

  // C layout: lane(c,n) vgpr r -> row = r + 8c, col = n
#pragma unroll
  for (int t = 0; t < 4; ++t) {
#pragma unroll
    for (int r = 0; r < 8; ++r) {
      const int srow = mt * 64 + wave * 16 + r + 8 * c;
      const int d = t * 16 + mn;
      qo[((long)bh * S_ + srow) * D_ + d]  = (_Float16)aq[t][r];
      ko[((long)bh * S_ + srow) * D_ + d]  = (_Float16)ak[t][r];
      vto[((long)bh * D_ + d) * S_ + srow] = (_Float16)av[t][r];
    }
  }
}

// ---------------- Flash attention (causal, online softmax) ----------------
// K/V tiles staged in LDS once per block (shared by all 4 waves).
__global__ __launch_bounds__(128) void flash_attn(
    const _Float16* __restrict__ q,    // [B*H][S][D]
    const _Float16* __restrict__ k,    // [B*H][S][D]
    const _Float16* __restrict__ vt,   // [B*H][D][S]
    _Float16* __restrict__ att)        // [B*S][H*D]
{
  __shared__ _Float16 Ks[64 * 72];     // [j_local][d], padded stride 72 halves
  __shared__ _Float16 Vs[64 * 72];     // [d][j_local]
  __shared__ _Float16 Ps[4][16 * 72];  // per-wave P tile
  const int tid  = threadIdx.x;
  const int wave = tid >> 5, lane = tid & 31;
  const int c = lane >> 4, mn = lane & 15;
  const int qi = blockIdx.x & 31;      // q row-tile
  const int bh = blockIdx.x >> 5;
  const int b = bh >> 4, h = bh & 15;
  const float scale = 0.125f;          // 1/sqrt(D)

  const int row0 = qi * 64 + wave * 16;

  // Q A-fragments for d=0..31 and d=32..63, kept in registers
  const _Float16* qrow = q + ((long)bh * S_ + row0 + mn) * D_;
  V16 a0, a1;
  a0.h[0] = *(const v8h*)&qrow[c * 16];
  a0.h[1] = *(const v8h*)&qrow[c * 16 + 8];
  a1.h[0] = *(const v8h*)&qrow[32 + c * 16];
  a1.h[1] = *(const v8h*)&qrow[32 + c * 16 + 8];

  v8f zf = {};
  v8f o[4];
  float mrow[8], lrow[8];
#pragma unroll
  for (int t = 0; t < 4; ++t) o[t] = zf;
#pragma unroll
  for (int r = 0; r < 8; ++r) { mrow[r] = -1e30f; lrow[r] = 0.f; }

  _Float16* pw = &Ps[wave][0];
  const int crow  = tid >> 1;          // cooperative-copy row 0..63
  const int chalf = (tid & 1) * 32;    // half-row of 32 halves

  for (int j0 = 0; j0 <= qi * 64; j0 += 64) {
    // ---- stage K (64x64) and V^T (64x64) tiles into LDS ----
    __syncthreads();
    const _Float16* kg = k  + ((long)bh * S_ + j0 + crow) * D_ + chalf;
    const _Float16* vg = vt + ((long)bh * D_ + crow) * S_ + j0 + chalf;
#pragma unroll
    for (int o8 = 0; o8 < 32; o8 += 8) {
      gl2lds_b128(kg + o8, &Ks[crow * 72 + chalf + o8]);
      gl2lds_b128(vg + o8, &Vs[crow * 72 + chalf + o8]);
    }
    wait_gl2lds();
    __syncthreads();

    // ---- S = Q K^T (4 column tiles of 16) ----
    v8f sc[4];
#pragma unroll
    for (int t = 0; t < 4; ++t) {
      const _Float16* kc = &Ks[(t * 16 + mn) * 72];   // col j = row of K tile
      V16 b0, b1;
      b0.h[0] = *(const v8h*)&kc[c * 16];
      b0.h[1] = *(const v8h*)&kc[c * 16 + 8];
      b1.h[0] = *(const v8h*)&kc[32 + c * 16];
      b1.h[1] = *(const v8h*)&kc[32 + c * 16 + 8];
      v8f s = wmma_f16(a0.v, b0.v, zf);
      s = wmma_f16(a1.v, b1.v, s);
      const int jcol = j0 + t * 16 + mn;
#pragma unroll
      for (int r = 0; r < 8; ++r) {
        const int irow = row0 + r + 8 * c;
        sc[t][r] = (jcol > irow) ? -1e30f : s[r] * scale;
      }
    }

    // ---- online softmax: half-wave (16-lane) reductions per row ----
    float alpha[8], rsum[8];
#pragma unroll
    for (int r = 0; r < 8; ++r) {
      float mv = fmaxf(fmaxf(sc[0][r], sc[1][r]), fmaxf(sc[2][r], sc[3][r]));
#pragma unroll
      for (int off = 1; off < 16; off <<= 1) mv = fmaxf(mv, __shfl_xor(mv, off));
      const float mnew = fmaxf(mrow[r], mv);
      alpha[r] = __expf(mrow[r] - mnew);
      mrow[r] = mnew;
      rsum[r] = 0.f;
    }
#pragma unroll
    for (int t = 0; t < 4; ++t)
#pragma unroll
      for (int r = 0; r < 8; ++r) {
        const float p = __expf(sc[t][r] - mrow[r]);
        sc[t][r] = p;
        rsum[r] += p;
      }
#pragma unroll
    for (int r = 0; r < 8; ++r) {
#pragma unroll
      for (int off = 1; off < 16; off <<= 1) rsum[r] += __shfl_xor(rsum[r], off);
      lrow[r] = lrow[r] * alpha[r] + rsum[r];
#pragma unroll
      for (int t = 0; t < 4; ++t) o[t][r] *= alpha[r];
    }

    // ---- spill P (C layout) to wave-private LDS, reload in A layout ----
#pragma unroll
    for (int t = 0; t < 4; ++t)
#pragma unroll
      for (int r = 0; r < 8; ++r)
        pw[(r + 8 * c) * 72 + t * 16 + mn] = (_Float16)sc[t][r];

    V16 pa0, pa1;
    pa0.h[0] = *(const v8h*)&pw[mn * 72 + c * 16];
    pa0.h[1] = *(const v8h*)&pw[mn * 72 + c * 16 + 8];
    pa1.h[0] = *(const v8h*)&pw[mn * 72 + 32 + c * 16];
    pa1.h[1] = *(const v8h*)&pw[mn * 72 + 32 + c * 16 + 8];

    // ---- O += P @ V  (V^T rows = B columns, contiguous in LDS) ----
#pragma unroll
    for (int t = 0; t < 4; ++t) {
      const _Float16* vc = &Vs[(t * 16 + mn) * 72];
      V16 b0, b1;
      b0.h[0] = *(const v8h*)&vc[c * 16];
      b0.h[1] = *(const v8h*)&vc[c * 16 + 8];
      b1.h[0] = *(const v8h*)&vc[32 + c * 16];
      b1.h[1] = *(const v8h*)&vc[32 + c * 16 + 8];
      o[t] = wmma_f16(pa0.v, b0.v, o[t]);
      o[t] = wmma_f16(pa1.v, b1.v, o[t]);
    }
  }

  // normalize and write concatenated heads
#pragma unroll
  for (int t = 0; t < 4; ++t)
#pragma unroll
    for (int r = 0; r < 8; ++r) {
      const int srow = row0 + r + 8 * c;
      att[((long)b * S_ + srow) * (H_ * D_) + h * D_ + t * 16 + mn] =
          (_Float16)(o[t][r] / lrow[r]);
    }
}

// ---------------- output projection: out = att @ Wp + bp ----------------
__global__ __launch_bounds__(128) void out_proj(
    const _Float16* __restrict__ att,  // [B*S][1024]
    const _Float16* __restrict__ wpt,  // [E][1024] (transposed Wp)
    const float* __restrict__ bp,      // [E]
    float* __restrict__ out)           // [B*S][E]
{
  __shared__ _Float16 As[64 * 40];
  const int tid  = threadIdx.x;
  const int wave = tid >> 5, lane = tid & 31;
  const int c = lane >> 4, mn = lane & 15;
  const int nt = blockIdx.x & 15;      // E/64
  const int mt = blockIdx.x >> 4;      // (B*S)/64
  const int K = H_ * D_;               // 1024

  const _Float16* abase = att + (long)mt * 64 * K;
  v8f zf = {};
  v8f acc[4];
#pragma unroll
  for (int t = 0; t < 4; ++t) acc[t] = zf;

  const int arow = tid >> 1;
  const int acol = (tid & 1) * 16;

  for (int k0 = 0; k0 < K; k0 += 32) {
    __syncthreads();
    gl2lds_b128(&abase[(long)arow * K + k0 + acol],     &As[arow * 40 + acol]);
    gl2lds_b128(&abase[(long)arow * K + k0 + acol + 8], &As[arow * 40 + acol + 8]);
    wait_gl2lds();
    __syncthreads();

    V16 a;
    a.h[0] = *(const v8h*)&As[(wave * 16 + mn) * 40 + c * 16];
    a.h[1] = *(const v8h*)&As[(wave * 16 + mn) * 40 + c * 16 + 8];

#pragma unroll
    for (int t = 0; t < 4; ++t) {
      const _Float16* wc = wpt + ((long)(nt * 64 + t * 16 + mn)) * K + k0 + c * 16;
      V16 bb;
      bb.h[0] = *(const v8h*)&wc[0];
      bb.h[1] = *(const v8h*)&wc[8];
      acc[t] = wmma_f16(a.v, bb.v, acc[t]);
    }
  }

#pragma unroll
  for (int t = 0; t < 4; ++t) {
    const int col = nt * 64 + t * 16 + mn;
    const float bias = bp[col];
#pragma unroll
    for (int r = 0; r < 8; ++r) {
      const int row = mt * 64 + wave * 16 + r + 8 * c;
      out[(long)row * E_ + col] = acc[t][r] + bias;
    }
  }
}

// ---------------- launcher ----------------
extern "C" void kernel_launch(void* const* d_in, const int* in_sizes, int n_in,
                              void* d_out, int out_size, void* d_ws, size_t ws_size,
                              hipStream_t stream) {
  (void)in_sizes; (void)n_in; (void)out_size; (void)ws_size;
  const float* x  = (const float*)d_in[0];
  const float* Wq = (const float*)d_in[1];
  const float* Wk = (const float*)d_in[2];
  const float* Wv = (const float*)d_in[3];
  const float* Wp = (const float*)d_in[4];
  const float* bp = (const float*)d_in[5];

  char* ws = (char*)d_ws;
  size_t off = 0;
  auto take = [&](size_t bytes) -> char* {
    char* p = ws + off;
    off += (bytes + 255) & ~size_t(255);
    return p;
  };

  const size_t nx   = (size_t)B_ * S_ * E_;
  const size_t nw   = (size_t)H_ * E_ * D_;
  const size_t nqkv = (size_t)B_ * H_ * S_ * D_;
  const size_t natt = (size_t)B_ * S_ * H_ * D_;

  _Float16* xh   = (_Float16*)take(nx * 2);
  _Float16* wqt  = (_Float16*)take(nw * 2);
  _Float16* wkt  = (_Float16*)take(nw * 2);
  _Float16* wvt  = (_Float16*)take(nw * 2);
  _Float16* wpt  = (_Float16*)take((size_t)E_ * H_ * D_ * 2);
  _Float16* qb   = (_Float16*)take(nqkv * 2);
  _Float16* kb   = (_Float16*)take(nqkv * 2);
  _Float16* vtb  = (_Float16*)take(nqkv * 2);
  _Float16* attb = (_Float16*)take(natt * 2);

  cvt_f32_to_f16<<<4096, 256, 0, stream>>>(x, xh, (int)nx);
  transpose_to_f16<<<2048, 256, 0, stream>>>(Wq, wqt, H_, E_, D_);
  transpose_to_f16<<<2048, 256, 0, stream>>>(Wk, wkt, H_, E_, D_);
  transpose_to_f16<<<2048, 256, 0, stream>>>(Wv, wvt, H_, E_, D_);
  transpose_to_f16<<<2048, 256, 0, stream>>>(Wp, wpt, 1, H_ * D_, E_);

  qkv_gemm<<<B_ * H_ * (S_ / 64), 128, 0, stream>>>(xh, wqt, wkt, wvt, qb, kb, vtb);
  flash_attn<<<B_ * H_ * (S_ / 64), 128, 0, stream>>>(qb, kb, vtb, attb);
  out_proj<<<(B_ * S_ / 64) * (E_ / 64), 128, 0, stream>>>(attb, wpt, bp, (float*)d_out);
}
